// CapsuleLayer_80358838108607
// MI455X (gfx1250) — matrix operand order repaired and checked by reading, not compiled, not run on hardware
//
#include <hip/hip_runtime.h>
#include <hip/hip_bf16.h>

typedef __bf16 v16bf __attribute__((ext_vector_type(16)));
typedef float  v8f   __attribute__((ext_vector_type(8)));

#define NB 64
#define NI 2048
#define ND 16
#define NN 32
#define NE 32

static __device__ __forceinline__ v8f wmma_bf16(v16bf a, v16bf b, v8f c) {
#if defined(__gfx1250__)
  // v_wmma_f32_16x16x32_bf16: D = A(16x32 bf16) * B(32x16 bf16) + C(16x16 f32)
  return __builtin_amdgcn_wmma_f32_16x16x32_bf16(false, a, false, b, (short)0, c,
                                                 false, false);
#else
  (void)a; (void)b;
  return c;
#endif
}

// ---------------------------------------------------------------------------
// Async global->LDS staging of the x tile (16 b-rows x CI*16 floats, padded
// row stride XSTR floats). Uses CDNA5 GLOBAL_LOAD_ASYNC_TO_LDS_B128 (ASYNCcnt)
// so the data never touches VGPRs. LDS destination operand is the wave-relative
// byte offset = low 32 bits of the flat shared-aperture address (ISA 10.2).
// ---------------------------------------------------------------------------
template <int CI>
static __device__ __forceinline__ void stage_x_async(
    float* xs, const float* __restrict__ x, int m0, int i0c, int tid) {
  constexpr int XSTR = CI * 16 + 4;
#if defined(__gfx1250__)
  const uint32_t xs_base = (uint32_t)(uintptr_t)xs;
#pragma unroll
  for (int idx = tid; idx < 16 * CI * 4; idx += 512) {
    const int b  = idx / (CI * 4);
    const int r4 = idx - b * (CI * 4);
    const float* gp = x + ((size_t)(m0 + b) * NI + i0c) * ND + r4 * 4;
    const uint32_t lo = xs_base + (uint32_t)(b * XSTR + r4 * 4) * 4u;
    asm volatile("global_load_async_to_lds_b128 %0, %1, off"
                 :: "v"(lo), "v"(gp) : "memory");
  }
  asm volatile("s_wait_asynccnt 0x0" ::: "memory");
#else
  for (int idx = tid; idx < 16 * CI * 16; idx += 512) {
    const int b = idx / (CI * 16);
    const int r = idx - b * (CI * 16);
    xs[b * XSTR + r] = x[((size_t)(m0 + b) * NI + i0c) * ND + r];
  }
#endif
}

// ---------------------------------------------------------------------------
// Convert W [I,N,D,E] f32 -> Wb bf16 in B-fragment layout for 16x16x32 WMMA.
// Fragment key: (ip = i/2, n, eh = e-half). Within a fragment, lane L holds 16
// bf16 (one v16bf): element j -> ki = j/8, d = 8*(L/16) + j%8, col = L%16, where
// k = ki*16 + d (ki = i within pair). So: i = 2*ip + j/8, d = 8*(L/16) + j%8, e = eh*16 + L%16.
// Flat: Wb[(((ip*NN + n)*2 + eh)*32 + lane)*16 + j]
// ---------------------------------------------------------------------------
__global__ void k_convert_w(const float* __restrict__ W, __bf16* __restrict__ Wb) {
  const int t = blockIdx.x * blockDim.x + threadIdx.x;
  const int lane = t & 31;
  const int eh   = (t >> 5) & 1;
  const int n    = (t >> 6) & 31;
  const int ip   = t >> 11;
  if (ip >= NI / 2) return;
  const int i = 2 * ip;
  const int e = eh * 16 + (lane & 15);
  const float* src = W + ((size_t)(i * NN + n) * ND + 8 * (lane >> 4)) * NE + e;  // step NE per d
  __bf16* dst = Wb + (size_t)t * 16;
#pragma unroll
  for (int j = 0; j < 16; ++j) dst[j] = (__bf16)src[(size_t)(j >> 3) * NN * ND * NE + (j & 7) * NE];
}

__global__ void k_zero(float* __restrict__ p, int n) {
  const int t = blockIdx.x * blockDim.x + threadIdx.x;
  if (t < n) p[t] = 0.0f;
}

// ---------------------------------------------------------------------------
// k_accum_s: s[b,n,e] += sum_i c[b,i,n] * u_hat[b,i,n,e], u_hat recomputed by
// WMMA. Block = 512 threads = 16 waves; wave w owns capsules n = 2w, 2w+1
// (each with 2 e-halves -> 4 f32 accum tiles). Grid = 4 m-tiles x i-chunks.
// A operand per n: A'[m, k=(ki,d)] = c[b,i0+ki,n] * x[b,i0+ki,d] (bf16).
// A 16-bit layout: lane L (m = L%16): j<8 -> (ki=0, d = 8*(L/16)+j),
//                  j>=8 -> (ki=1, d = 8*(L/16)+j-8).
// ---------------------------------------------------------------------------
template <int CI>
__global__ __launch_bounds__(512, 1) void k_accum_s(
    const float* __restrict__ x, const __bf16* __restrict__ Wb,
    const float* __restrict__ cbuf, float cuni, float* __restrict__ s) {
  constexpr int XSTR = CI * 16 + 4;  // pad to avoid LDS bank conflicts
  extern __shared__ float xs[];      // [16][XSTR]
  const int mtile = blockIdx.x & 3;
  const int chunk = blockIdx.x >> 2;
  const int m0  = mtile * 16;
  const int i0c = chunk * CI;
  const int tid = threadIdx.x;
  const int wave = tid >> 5;
  const int lane = tid & 31;

  stage_x_async<CI>(xs, x, m0, i0c, tid);
  __syncthreads();

  const int n0 = wave * 2;
  const int m = lane & 15;
  const int dbase = (lane >> 4) * 8;

  v8f acc00 = {}; v8f acc01 = {}; v8f acc10 = {}; v8f acc11 = {};

  for (int st = 0; st < CI / 2; ++st) {
    const int il0 = 2 * st, il1 = il0 + 1;
    const int i0 = i0c + il0;
    float xa[8], xb[8];
#pragma unroll
    for (int j = 0; j < 8; ++j) {
      xa[j] = xs[m * XSTR + il0 * ND + dbase + j];
      xb[j] = xs[m * XSTR + il1 * ND + dbase + j];
    }
    float c00, c01, c10, c11;  // c[i0,n0], c[i1,n0], c[i0,n1], c[i1,n1]
    if (cbuf) {
      const size_t cb = ((size_t)(m0 + m) * NI + i0) * NN;
      c00 = cbuf[cb + n0];
      c01 = cbuf[cb + NN + n0];
      c10 = cbuf[cb + n0 + 1];
      c11 = cbuf[cb + NN + n0 + 1];
    } else {
      c00 = c01 = c10 = c11 = cuni;  // round 0: uniform softmax(0) = 1/32
    }
    v16bf A0, A1;
#pragma unroll
    for (int j = 0; j < 8; ++j) {
      A0[j]     = (__bf16)(xa[j] * c00);
      A0[j + 8] = (__bf16)(xb[j] * c01);
      A1[j]     = (__bf16)(xa[j] * c10);
      A1[j + 8] = (__bf16)(xb[j] * c11);
    }
    const size_t ip = (size_t)(i0 >> 1);
    {
      const __bf16* fb = Wb + (((ip * NN + n0) * 2) * 32 + lane) * 16;
#if defined(__gfx1250__)
      __builtin_prefetch(fb + 32768, 0, 0);  // next ip, same n -> global_prefetch
#endif
      v16bf B0 = *(const v16bf*)fb;
      v16bf B1 = *(const v16bf*)(fb + 512);
      acc00 = wmma_bf16(A0, B0, acc00);
      acc01 = wmma_bf16(A0, B1, acc01);
    }
    {
      const __bf16* fb = Wb + (((ip * NN + n0 + 1) * 2) * 32 + lane) * 16;
      v16bf B0 = *(const v16bf*)fb;
      v16bf B1 = *(const v16bf*)(fb + 512);
      acc10 = wmma_bf16(A1, B0, acc10);
      acc11 = wmma_bf16(A1, B1, acc11);
    }
  }

  // C layout: reg r, lanes 0-15: (m=r, e=lane); lanes 16-31: (m=r+8, e=lane-16)
  const int bo = m0 + 8 * (lane >> 4);
  const int el = lane & 15;
#pragma unroll
  for (int r = 0; r < 8; ++r) {
    float* sp = s + ((size_t)(bo + r) * NN + n0) * NE;
    atomicAdd(sp + el,           acc00[r]);
    atomicAdd(sp + 16 + el,      acc01[r]);
    atomicAdd(sp + NE + el,      acc10[r]);
    atomicAdd(sp + NE + 16 + el, acc11[r]);
  }
}

// ---------------------------------------------------------------------------
// k_squash: v = squash(s) over e (32 lanes == one capsule, wave32). Re-zeros s
// for the next accumulation round.
// ---------------------------------------------------------------------------
__global__ void k_squash(float* __restrict__ s, float* __restrict__ vout) {
  const int g = blockIdx.x * 8 + (threadIdx.x >> 5);  // g = b*NN + n
  const int lane = threadIdx.x & 31;
  if (g >= NB * NN) return;
  const size_t idx = (size_t)g * NE + lane;
  const float sv = s[idx];
  float s2 = sv * sv;
#pragma unroll
  for (int o = 16; o >= 1; o >>= 1) s2 += __shfl_xor(s2, o, 32);
  const float scale = s2 / (1.0f + s2) * rsqrtf(s2 + 1e-7f);
  vout[idx] = scale * sv;
  s[idx] = 0.0f;
}

// ---------------------------------------------------------------------------
// k_route: logit[b,i,n] = (Lold) + sum_e u_hat[b,i,n,e]*v[b,n,e]; then softmax
// over n -> cbuf, store updated logits -> Lnew. u_hat recomputed per single i
// via WMMA with half-zeroed A; dot over e via shfl_xor within 16-lane halves.
// ---------------------------------------------------------------------------
template <int CI>
__global__ __launch_bounds__(512, 1) void k_route(
    const float* __restrict__ x, const __bf16* __restrict__ Wb,
    const float* __restrict__ vprev, const float* __restrict__ Lold,
    float* __restrict__ Lnew, float* __restrict__ cbuf) {
  constexpr int XSTR = CI * 16 + 4;
  extern __shared__ float smem[];
  float* xs = smem;                // 16*XSTR floats
  float* dl = smem + 16 * XSTR;    // [CI][16][NN] dots
  const int mtile = blockIdx.x & 3;
  const int chunk = blockIdx.x >> 2;
  const int m0  = mtile * 16;
  const int i0c = chunk * CI;
  const int tid = threadIdx.x;
  const int wave = tid >> 5;
  const int lane = tid & 31;

  stage_x_async<CI>(xs, x, m0, i0c, tid);
  __syncthreads();

  const int n0 = wave * 2;
  const int m = lane & 15;
  const int dbase = (lane >> 4) * 8;

  // Preload v fragments matching the C-tile layout: vv[nn][eh][r]
  float vv[2][2][8];
#pragma unroll
  for (int nn = 0; nn < 2; ++nn) {
#pragma unroll
    for (int r = 0; r < 8; ++r) {
      const int b = m0 + r + 8 * (lane >> 4);
      const float* vp = vprev + ((size_t)b * NN + n0 + nn) * NE;
      vv[nn][0][r] = vp[lane & 15];
      vv[nn][1][r] = vp[16 + (lane & 15)];
    }
  }

  for (int st = 0; st < CI / 2; ++st) {
    const int il0 = 2 * st;
    const int i0 = i0c + il0;
    v16bf A0, A1;  // A0: i0 in low-K half, A1: i1 in high-K half (rest zero)
#pragma unroll
    for (int j = 0; j < 8; ++j) {
      A0[j]     = (__bf16)xs[m * XSTR + il0 * ND + dbase + j];
      A0[j + 8] = (__bf16)0.0f;
      A1[j]     = (__bf16)0.0f;
      A1[j + 8] = (__bf16)xs[m * XSTR + (il0 + 1) * ND + dbase + j];
    }
#pragma unroll
    for (int nn = 0; nn < 2; ++nn) {
      const __bf16* fb =
          Wb + ((((size_t)(i0 >> 1)) * NN + n0 + nn) * 2 * 32 + lane) * 16;
      v16bf B0 = *(const v16bf*)fb;
      v16bf B1 = *(const v16bf*)(fb + 512);
#pragma unroll
      for (int ii = 0; ii < 2; ++ii) {
        v8f z0 = {}; v8f z1 = {};
        v8f T0 = wmma_bf16(ii ? A1 : A0, B0, z0);
        v8f T1 = wmma_bf16(ii ? A1 : A0, B1, z1);
#pragma unroll
        for (int r = 0; r < 8; ++r) {
          float p = T0[r] * vv[nn][0][r] + T1[r] * vv[nn][1][r];
          p += __shfl_xor(p, 1, 32);
          p += __shfl_xor(p, 2, 32);
          p += __shfl_xor(p, 4, 32);
          p += __shfl_xor(p, 8, 32);  // sum over 16 e-lanes within half
          if ((lane & 15) == r)
            dl[((il0 + ii) * 16 + (r + 8 * (lane >> 4))) * NN + n0 + nn] = p;
        }
      }
    }
  }
  __syncthreads();

  // Softmax over n per (b, i); emit routing coefficients c and updated logits.
  for (int pr = tid; pr < 16 * CI; pr += 512) {
    const int bl = pr & 15;
    const int il = pr >> 4;
    const size_t gb = ((size_t)(m0 + bl) * NI + (i0c + il)) * NN;
    float lg[NN];
    float mx = -3.0e38f;
#pragma unroll
    for (int n = 0; n < NN; ++n) {
      float lv = dl[(il * 16 + bl) * NN + n];
      if (Lold) lv += Lold[gb + n];
      Lnew[gb + n] = lv;
      lg[n] = lv;
      mx = fmaxf(mx, lv);
    }
    float ssum = 0.0f;
#pragma unroll
    for (int n = 0; n < NN; ++n) { const float e = __expf(lg[n] - mx); lg[n] = e; ssum += e; }
    const float inv = 1.0f / ssum;
#pragma unroll
    for (int n = 0; n < NN; ++n) cbuf[gb + n] = lg[n] * inv;
  }
}

extern "C" void kernel_launch(void* const* d_in, const int* in_sizes, int n_in,
                              void* d_out, int out_size, void* d_ws, size_t ws_size,
                              hipStream_t stream) {
  (void)in_sizes; (void)n_in; (void)out_size; (void)ws_size;
  const float* x = (const float*)d_in[0];
  const float* W = (const float*)d_in[1];
  float* out = (float*)d_out;

  char* ws = (char*)d_ws;
  __bf16* Wb = (__bf16*)ws;
  size_t off = (size_t)NI * NN * ND * NE * sizeof(__bf16);       // 64 MiB
  float* cbuf = (float*)(ws + off); off += (size_t)NB * NI * NN * sizeof(float);
  float* Lbuf = (float*)(ws + off); off += (size_t)NB * NI * NN * sizeof(float);
  float* sbuf = (float*)(ws + off); off += (size_t)NB * NN * NE * sizeof(float);
  float* vbuf = (float*)(ws + off);

  // One-time (per call) bf16 fragment conversion of W.
  k_convert_w<<<8192, 256, 0, stream>>>(W, Wb);
  k_zero<<<(NB * NN * NE + 255) / 256, 256, 0, stream>>>(sbuf, NB * NN * NE);

  constexpr int CI_A = 32, CI_B = 16;
  const dim3 gA(4 * (NI / CI_A));   // 256 workgroups
  const size_t ldsA = (size_t)16 * (CI_A * 16 + 4) * sizeof(float);
  const dim3 gB(4 * (NI / CI_B));   // 512 workgroups
  const size_t ldsB =
      ((size_t)16 * (CI_B * 16 + 4) + (size_t)CI_B * 16 * NN) * sizeof(float);

  // Round 0 (uniform c = 1/32)
  k_accum_s<CI_A><<<gA, 512, ldsA, stream>>>(x, Wb, nullptr, 1.0f / 32.0f, sbuf);
  k_squash<<<256, 256, 0, stream>>>(sbuf, vbuf);
  k_route<CI_B><<<gB, 512, ldsB, stream>>>(x, Wb, vbuf, nullptr, Lbuf, cbuf);
  // Round 1
  k_accum_s<CI_A><<<gA, 512, ldsA, stream>>>(x, Wb, cbuf, 0.0f, sbuf);
  k_squash<<<256, 256, 0, stream>>>(sbuf, vbuf);
  k_route<CI_B><<<gB, 512, ldsB, stream>>>(x, Wb, vbuf, Lbuf, Lbuf, cbuf);
  // Round 2
  k_accum_s<CI_A><<<gA, 512, ldsA, stream>>>(x, Wb, cbuf, 0.0f, sbuf);
  k_squash<<<256, 256, 0, stream>>>(sbuf, out);
}
